// RGLRU_55284819034487
// MI455X (gfx1250) — compile-verified, matrix operand order checked by reading, not run on hardware
//
#include <hip/hip_runtime.h>
#include <hip/hip_bf16.h>
#include <math.h>

// ---------------- problem sizes ----------------
#define D_MODEL 1024
#define D_STATE 1024
#define BATCH   4
#define SEQ     4096
#define MTOT    (BATCH * SEQ)   // 16384 rows

// ---------------- GEMM tiling ------------------
#define BM 64        // block tile M (2 waves of 32)
#define BN 128       // block tile N (4 waves of 32)
#define KT 32        // K per WMMA step (bf16)
#define APITCH 40    // bf16 elems per LDS row (80B, 16B-aligned rows)
#define BPITCH 40

// ---------------- scan chunking ----------------
#define CHUNKS 32
#define CLEN   (SEQ / CHUNKS)   // 128

typedef __attribute__((ext_vector_type(16))) __bf16 v16bf;
typedef __attribute__((ext_vector_type(8)))  __bf16 v8bf;
typedef __attribute__((ext_vector_type(8)))  float  v8f;
typedef __attribute__((ext_vector_type(4)))  float  v4f;

// float -> bf16, round-to-nearest-even
__device__ __forceinline__ __bf16 f2bf(float f) {
    unsigned u = __builtin_bit_cast(unsigned, f);
    u += 0x7FFFu + ((u >> 16) & 1u);
    unsigned short h = (unsigned short)(u >> 16);
    return __builtin_bit_cast(__bf16, h);
}

__device__ __forceinline__ float sigm(float x) {
    return 1.0f / (1.0f + __expf(-x));
}

// LDS byte offset of a __shared__ object: flat LDS-aperture address low 32 bits
// equal the offset within the workgroup's LDS allocation (ISA 10.2).
__device__ __forceinline__ unsigned lds_off(const void* p) {
    return (unsigned)(uintptr_t)p;
}

// CDNA5 async copy: 16 bytes global -> LDS per lane, tracked by ASYNCcnt.
__device__ __forceinline__ void async_ld_b128(unsigned lds, const void* gaddr) {
    asm volatile("global_load_async_to_lds_b128 %0, %1, off"
                 :: "v"(lds), "v"((unsigned long long)(uintptr_t)gaddr)
                 : "memory");
}
__device__ __forceinline__ void wait_async0() {
    asm volatile("s_wait_asynccnt 0" ::: "memory");
}

// Load one 16x32 bf16 fragment from an LDS tile stored row-major with `pitch`.
// Layout per CDNA5 ISA (16-bit A 16x32): lanes 0-15 row M=lane hold K=0..7 &
// K=16..23; lanes 16-31 row M=lane-16 hold K=8..15 & K=24..31.
__device__ __forceinline__ v16bf load_frag(const __bf16* base, int pitch, int lane) {
    int r    = lane & 15;
    int kOff = (lane >> 4) ? 8 : 0;
    const __bf16* p = base + r * pitch + kOff;
    v8bf x0 = *(const v8bf*)(p);
    v8bf x1 = *(const v8bf*)(p + 16);
    v16bf f;
#pragma unroll
    for (int i = 0; i < 8; ++i) { f[i] = x0[i]; f[8 + i] = x1[i]; }
    return f;
}

__device__ __forceinline__ v8f wmma_bf16(v16bf a, v16bf b, v8f c) {
    return __builtin_amdgcn_wmma_f32_16x16x32_bf16(
        /*neg_a=*/false, a, /*neg_b=*/false, b,
        /*c_mod=*/(short)0, c, /*reuse_a=*/false, /*reuse_b=*/false);
}

// =====================================================================
// Preprocessing: fp32 -> bf16 copy (x), and fp32 -> bf16 transpose (W).
// =====================================================================
__global__ __launch_bounds__(256) void cvt_bf16(const float* __restrict__ src,
                                                __bf16* __restrict__ dst, int n) {
    int i = (blockIdx.x * blockDim.x + threadIdx.x) * 8;
    if (i >= n) return;
    v4f f0 = *(const v4f*)(src + i);
    v4f f1 = *(const v4f*)(src + i + 4);
    v8bf o;
#pragma unroll
    for (int j = 0; j < 4; ++j) { o[j] = f2bf(f0[j]); o[4 + j] = f2bf(f1[j]); }
    *(v8bf*)(dst + i) = o;
}

// Wt[n][k] = bf16(W[k][n]), both 1024x1024. 64x64 tiles via LDS.
__global__ __launch_bounds__(256) void cvt_transpose_bf16(const float* __restrict__ W,
                                                          __bf16* __restrict__ Wt) {
    __shared__ float tile[64][65];
    const int t  = threadIdx.x;
    const int k0 = blockIdx.y * 64;
    const int n0 = blockIdx.x * 64;
#pragma unroll
    for (int i = 0; i < 16; ++i) {
        int id = t + 256 * i;           // 0..4095
        int kk = id >> 6, nn = id & 63;
        tile[kk][nn] = W[(size_t)(k0 + kk) * D_STATE + n0 + nn];
    }
    __syncthreads();
#pragma unroll
    for (int i = 0; i < 16; ++i) {
        int id = t + 256 * i;
        int nn = id >> 6, kk = id & 63;
        Wt[(size_t)(n0 + nn) * D_MODEL + k0 + kk] = f2bf(tile[kk][nn]);
    }
}

// =====================================================================
// Kernel 1: fused xb@{Wt_in,Wt_r,Wt_i}^T + bias + sigmoid gating.
// Double-buffered async global->LDS staging overlapped with WMMA compute.
// =====================================================================
__global__ __launch_bounds__(256) void rglru_proj_gate(
    const __bf16* __restrict__ xb,
    const __bf16* __restrict__ Wtin, const float* __restrict__ bin,
    const __bf16* __restrict__ Wtr,  const float* __restrict__ br,
    const __bf16* __restrict__ Wti,  const float* __restrict__ bi,
    const float* __restrict__ log_a,
    float* __restrict__ at_out, float* __restrict__ bt_out)
{
    __shared__ __align__(16) __bf16 As[2][BM * APITCH];
    __shared__ __align__(16) __bf16 Bs[2][3][BN * BPITCH];

    const int tid   = threadIdx.x;
    const int lane  = tid & 31;
    const int wave  = tid >> 5;
    const int waveM = wave >> 2;   // 0..1
    const int waveN = wave & 3;    // 0..3
    const int m0 = blockIdx.y * BM;
    const int n0 = blockIdx.x * BN;

    const __bf16* Wp[3] = { Wtin, Wtr, Wti };
    unsigned asBase[2], bsBase[2][3];
#pragma unroll
    for (int s = 0; s < 2; ++s) {
        asBase[s] = lds_off(As[s]);
#pragma unroll
        for (int w = 0; w < 3; ++w) bsBase[s][w] = lds_off(Bs[s][w]);
    }

    v8f zero = {};
    v8f acc[3][2][2];
#pragma unroll
    for (int w = 0; w < 3; ++w)
#pragma unroll
        for (int i = 0; i < 2; ++i)
#pragma unroll
            for (int j = 0; j < 2; ++j) acc[w][i][j] = zero;

    // per-thread staging coordinates (constant across K loop)
    const int aRow = tid >> 2;            // 0..63
    const int aOff = (tid & 3) * 16;      // byte offset within 64B row

    auto stage = [&](int buf, int k0) {
        // A tile: 64 rows x 64B, 1 async b128 per thread
        async_ld_b128(asBase[buf] + aRow * (APITCH * 2) + aOff,
                      (const char*)(xb + (size_t)(m0 + aRow) * D_MODEL + k0) + aOff);
        // B tiles: 3 x (128 rows x 64B), 2 async b128 per thread each
#pragma unroll
        for (int w = 0; w < 3; ++w) {
#pragma unroll
            for (int l = 0; l < 2; ++l) {
                int id  = tid + 256 * l;      // 0..511
                int row = id >> 2;            // 0..127
                int off = (id & 3) * 16;
                async_ld_b128(bsBase[buf][w] + row * (BPITCH * 2) + off,
                              (const char*)(Wp[w] + (size_t)(n0 + row) * D_MODEL + k0) + off);
            }
        }
    };

    // prologue
    stage(0, 0);
    wait_async0();
    __syncthreads();

    for (int k0 = 0, cur = 0; k0 < D_MODEL; k0 += KT, cur ^= 1) {
        // kick off next tile's async copies; they land during the WMMAs below
        if (k0 + KT < D_MODEL) stage(cur ^ 1, k0 + KT);

        const __bf16* Asb = As[cur];
        v16bf aF[2];
        aF[0] = load_frag(Asb + (waveM * 32 +  0) * APITCH, APITCH, lane);
        aF[1] = load_frag(Asb + (waveM * 32 + 16) * APITCH, APITCH, lane);
#pragma unroll
        for (int w = 0; w < 3; ++w) {
            const __bf16* Bsb = Bs[cur][w];
            v16bf bF0 = load_frag(Bsb + (waveN * 32 +  0) * BPITCH, BPITCH, lane);
            v16bf bF1 = load_frag(Bsb + (waveN * 32 + 16) * BPITCH, BPITCH, lane);
            acc[w][0][0] = wmma_bf16(aF[0], bF0, acc[w][0][0]);
            acc[w][0][1] = wmma_bf16(aF[0], bF1, acc[w][0][1]);
            acc[w][1][0] = wmma_bf16(aF[1], bF0, acc[w][1][0]);
            acc[w][1][1] = wmma_bf16(aF[1], bF1, acc[w][1][1]);
        }

        wait_async0();     // this wave's copies for the next buffer are done
        __syncthreads();   // everyone finished compute(cur) and staging(next)
    }

    // ---- epilogue: bias + sigmoid gating, store a_t / b_t (fp32) ----
    const int colLane = lane & 15;
    const int rowSel  = (lane >> 4) * 8;
#pragma unroll
    for (int ni = 0; ni < 2; ++ni) {
        int n = n0 + waveN * 32 + ni * 16 + colLane;
        float a_n   = sigm(log_a[n]);
        float bin_n = bin[n];
        float br_n  = br[n];
        float bi_n  = bi[n];
#pragma unroll
        for (int mi = 0; mi < 2; ++mi) {
            int rbase = m0 + waveM * 32 + mi * 16 + rowSel;
            v8f p  = acc[0][mi][ni];
            v8f rg = acc[1][mi][ni];
            v8f ig = acc[2][mi][ni];
#pragma unroll
            for (int j = 0; j < 8; ++j) {
                int m = rbase + j;
                float pv  = p[j] + bin_n;
                float rv  = sigm(rg[j] + br_n);
                float iv  = sigm(ig[j] + bi_n);
                float atv = a_n * rv;
                float btv = (1.0f - atv) * (iv * pv);
                size_t idx = (size_t)m * D_STATE + n;
                at_out[idx] = atv;
                bt_out[idx] = btv;
            }
        }
    }
}

// =====================================================================
// Kernel 2: per-chunk scan summaries. thread = (b, c, n).
// =====================================================================
__global__ __launch_bounds__(256) void rglru_scan_chunk(
    const float* __restrict__ at, const float* __restrict__ bt,
    float* __restrict__ chunkA, float* __restrict__ chunkH)
{
    int idx = blockIdx.x * blockDim.x + threadIdx.x;
    if (idx >= BATCH * CHUNKS * D_STATE) return;
    int n = idx % D_STATE;
    int c = (idx / D_STATE) % CHUNKS;
    int b = idx / (D_STATE * CHUNKS);
    size_t base = ((size_t)b * SEQ + (size_t)c * CLEN) * D_STATE + n;
    float h = 0.0f, A = 1.0f;
#pragma unroll 4
    for (int t = 0; t < CLEN; ++t) {
        float a  = at[base + (size_t)t * D_STATE];
        float bb = bt[base + (size_t)t * D_STATE];
        h = fmaf(a, h, bb);
        A *= a;
    }
    chunkA[idx] = A;
    chunkH[idx] = h;
}

// =====================================================================
// Kernel 3: sequential combine over chunk summaries per (b, n).
// =====================================================================
__global__ __launch_bounds__(256) void rglru_scan_combine(
    const float* __restrict__ chunkA, const float* __restrict__ chunkH,
    float* __restrict__ hstart)
{
    int idx = blockIdx.x * blockDim.x + threadIdx.x;
    if (idx >= BATCH * D_STATE) return;
    int n = idx % D_STATE;
    int b = idx / D_STATE;
    float h = 0.0f;
#pragma unroll
    for (int c = 0; c < CHUNKS; ++c) {
        size_t cidx = ((size_t)b * CHUNKS + c) * D_STATE + n;
        hstart[cidx] = h;
        h = fmaf(chunkA[cidx], h, chunkH[cidx]);
    }
}

// =====================================================================
// Kernel 4: re-scan each chunk with known h_start, write h as bf16
// (consumed only by the bf16 output GEMM).
// =====================================================================
__global__ __launch_bounds__(256) void rglru_scan_apply(
    const float* __restrict__ at, const float* __restrict__ bt,
    const float* __restrict__ hstart, __bf16* __restrict__ Hb)
{
    int idx = blockIdx.x * blockDim.x + threadIdx.x;
    if (idx >= BATCH * CHUNKS * D_STATE) return;
    int n = idx % D_STATE;
    int c = (idx / D_STATE) % CHUNKS;
    int b = idx / (D_STATE * CHUNKS);
    size_t base = ((size_t)b * SEQ + (size_t)c * CLEN) * D_STATE + n;
    float h = hstart[idx];
#pragma unroll 4
    for (int t = 0; t < CLEN; ++t) {
        float a  = at[base + (size_t)t * D_STATE];
        float bb = bt[base + (size_t)t * D_STATE];
        h = fmaf(a, h, bb);
        Hb[base + (size_t)t * D_STATE] = f2bf(h);
    }
}

// =====================================================================
// Kernel 5: output GEMM  out = H @ W_out + b_out  (bf16 WMMA,
// double-buffered async LDS staging).
// =====================================================================
__global__ __launch_bounds__(256) void rglru_out_gemm(
    const __bf16* __restrict__ Hb,
    const __bf16* __restrict__ Wtout, const float* __restrict__ bout,
    float* __restrict__ out)
{
    __shared__ __align__(16) __bf16 As[2][BM * APITCH];
    __shared__ __align__(16) __bf16 Bs[2][BN * BPITCH];

    const int tid   = threadIdx.x;
    const int lane  = tid & 31;
    const int wave  = tid >> 5;
    const int waveM = wave >> 2;
    const int waveN = wave & 3;
    const int m0 = blockIdx.y * BM;
    const int n0 = blockIdx.x * BN;

    unsigned asBase[2] = { lds_off(As[0]), lds_off(As[1]) };
    unsigned bsBase[2] = { lds_off(Bs[0]), lds_off(Bs[1]) };

    v8f zero = {};
    v8f acc[2][2];
#pragma unroll
    for (int i = 0; i < 2; ++i)
#pragma unroll
        for (int j = 0; j < 2; ++j) acc[i][j] = zero;

    const int aRow = tid >> 2;
    const int aOff = (tid & 3) * 16;

    auto stage = [&](int buf, int k0) {
        async_ld_b128(asBase[buf] + aRow * (APITCH * 2) + aOff,
                      (const char*)(Hb + (size_t)(m0 + aRow) * D_STATE + k0) + aOff);
#pragma unroll
        for (int l = 0; l < 2; ++l) {
            int id  = tid + 256 * l;
            int row = id >> 2;
            int off = (id & 3) * 16;
            async_ld_b128(bsBase[buf] + row * (BPITCH * 2) + off,
                          (const char*)(Wtout + (size_t)(n0 + row) * D_STATE + k0) + off);
        }
    };

    stage(0, 0);
    wait_async0();
    __syncthreads();

    for (int k0 = 0, cur = 0; k0 < D_STATE; k0 += KT, cur ^= 1) {
        if (k0 + KT < D_STATE) stage(cur ^ 1, k0 + KT);

        v16bf aF0 = load_frag(As[cur] + (waveM * 32 +  0) * APITCH, APITCH, lane);
        v16bf aF1 = load_frag(As[cur] + (waveM * 32 + 16) * APITCH, APITCH, lane);
        v16bf bF0 = load_frag(Bs[cur] + (waveN * 32 +  0) * BPITCH, BPITCH, lane);
        v16bf bF1 = load_frag(Bs[cur] + (waveN * 32 + 16) * BPITCH, BPITCH, lane);
        acc[0][0] = wmma_bf16(aF0, bF0, acc[0][0]);
        acc[0][1] = wmma_bf16(aF0, bF1, acc[0][1]);
        acc[1][0] = wmma_bf16(aF1, bF0, acc[1][0]);
        acc[1][1] = wmma_bf16(aF1, bF1, acc[1][1]);

        wait_async0();
        __syncthreads();
    }

    const int colLane = lane & 15;
    const int rowSel  = (lane >> 4) * 8;
#pragma unroll
    for (int ni = 0; ni < 2; ++ni) {
        int n = n0 + waveN * 32 + ni * 16 + colLane;
        float b_n = bout[n];
#pragma unroll
        for (int mi = 0; mi < 2; ++mi) {
            int rbase = m0 + waveM * 32 + mi * 16 + rowSel;
            v8f cacc = acc[mi][ni];
#pragma unroll
            for (int j = 0; j < 8; ++j) {
                int m = rbase + j;
                out[(size_t)m * D_MODEL + n] = cacc[j] + b_n;
            }
        }
    }
}

// =====================================================================
// Host launcher
// =====================================================================
extern "C" void kernel_launch(void* const* d_in, const int* in_sizes, int n_in,
                              void* d_out, int out_size, void* d_ws, size_t ws_size,
                              hipStream_t stream) {
    (void)in_sizes; (void)n_in; (void)out_size; (void)ws_size;

    const float* x     = (const float*)d_in[0];
    const float* Win   = (const float*)d_in[1];
    const float* bin   = (const float*)d_in[2];
    const float* Wr    = (const float*)d_in[3];
    const float* br    = (const float*)d_in[4];
    const float* Wi    = (const float*)d_in[5];
    const float* bi    = (const float*)d_in[6];
    const float* Wout  = (const float*)d_in[7];
    const float* bout  = (const float*)d_in[8];
    const float* log_a = (const float*)d_in[9];
    float* out = (float*)d_out;

    // workspace layout
    char* ws = (char*)d_ws;
    const size_t big   = (size_t)MTOT * D_STATE * sizeof(float);       // 64 MB
    const size_t bigBF = (size_t)MTOT * D_STATE * sizeof(short);       // 32 MB
    const size_t wBF   = (size_t)D_MODEL * D_STATE * sizeof(short);    //  2 MB
    const size_t sml   = (size_t)BATCH * CHUNKS * D_STATE * sizeof(float);

    float*  at     = (float*)(ws);
    float*  bt     = (float*)(ws + big);
    __bf16* xb     = (__bf16*)(ws + 2 * big);
    __bf16* Hb     = (__bf16*)(ws + 2 * big + bigBF);
    __bf16* Wtin   = (__bf16*)(ws + 2 * big + 2 * bigBF);
    __bf16* Wtr    = (__bf16*)(ws + 2 * big + 2 * bigBF + 1 * wBF);
    __bf16* Wti    = (__bf16*)(ws + 2 * big + 2 * bigBF + 2 * wBF);
    __bf16* Wtout  = (__bf16*)(ws + 2 * big + 2 * bigBF + 3 * wBF);
    float*  chunkA = (float*)(ws + 2 * big + 2 * bigBF + 4 * wBF);
    float*  chunkH = (float*)(ws + 2 * big + 2 * bigBF + 4 * wBF + sml);
    float*  hstart = (float*)(ws + 2 * big + 2 * bigBF + 4 * wBF + 2 * sml);

    // 0) one-time (per launch) bf16 conversion / weight transposition
    int nx = MTOT * D_MODEL;
    cvt_bf16<<<nx / (256 * 8), 256, 0, stream>>>(x, xb, nx);
    dim3 gT(D_MODEL / 64, D_STATE / 64);   // (16,16)
    cvt_transpose_bf16<<<gT, 256, 0, stream>>>(Win,  Wtin);
    cvt_transpose_bf16<<<gT, 256, 0, stream>>>(Wr,   Wtr);
    cvt_transpose_bf16<<<gT, 256, 0, stream>>>(Wi,   Wti);
    cvt_transpose_bf16<<<gT, 256, 0, stream>>>(Wout, Wtout);

    // 1) fused projections + gating
    dim3 grid1(D_STATE / BN, MTOT / BM);   // (8, 256)
    rglru_proj_gate<<<grid1, 256, 0, stream>>>(xb, Wtin, bin, Wtr, br, Wti, bi,
                                               log_a, at, bt);

    // 2..4) chunked associative scan
    int nChunkThreads = BATCH * CHUNKS * D_STATE;   // 131072
    rglru_scan_chunk<<<(nChunkThreads + 255) / 256, 256, 0, stream>>>(at, bt, chunkA, chunkH);
    int nChan = BATCH * D_STATE;                    // 4096
    rglru_scan_combine<<<(nChan + 255) / 256, 256, 0, stream>>>(chunkA, chunkH, hstart);
    rglru_scan_apply<<<(nChunkThreads + 255) / 256, 256, 0, stream>>>(at, bt, hstart, Hb);

    // 5) output projection
    dim3 grid5(D_MODEL / BN, MTOT / BM);   // (8, 256)
    rglru_out_gemm<<<grid5, 256, 0, stream>>>(Hb, Wtout, bout, out);
}
